// BasicQuantumAttention_73126113181742
// MI455X (gfx1250) — matrix operand order repaired and checked by reading, not compiled, not run on hardware
//
#include <hip/hip_runtime.h>
#include <hip/hip_bf16.h>
#include <math.h>

// ---------------------------------------------------------------------------
// BasicQuantumAttention for gfx1250 (MI455X, wave32, WMMA).
// Shapes fixed by the reference: B=2, L=512, D=128.
//
// Key transform: interference = sum_d cos(qp_d - kp_d)
//                             = dot(cos qp, cos kp) + dot(sin qp, sin kp)
//   -> a K=256 GEMM on v_wmma_f32_16x16x32_f16 (f32 accumulate).
// coherence = exp(-L1(qp,kp)) stays a VALU pass fused into the score tile.
// ---------------------------------------------------------------------------

typedef __attribute__((ext_vector_type(16))) _Float16 v16h;
typedef __attribute__((ext_vector_type(8)))  float    v8f;

#define BB 2
#define LL 512
#define DD 128
#define BL (BB * LL)

__constant__ const float kLnEps  = 1e-5f;
__constant__ const float kPhEps  = 1e-8f;

__device__ __forceinline__ v8f wmma16x16x32(const v16h& a, const v16h& b, const v8f& c) {
  // 8 args: (neg_a, A, neg_b, B, c_mod, C, reuse_a, reuse_b)
  return __builtin_amdgcn_wmma_f32_16x16x32_f16(false, a, false, b, (short)0, c,
                                                false, false);
}

// ---------------------------------------------------------------------------
// K1: Y = LayerNorm(X @ W^T) * g + b, rows handled 16 at a time per wave.
//   X:[1024,128], W:[128,128] (out = x @ W.T -> B[k][n] = W[n][k]).
// ---------------------------------------------------------------------------
__global__ __launch_bounds__(32)
void k_proj_ln(const float* __restrict__ X, const float* __restrict__ W,
               const float* __restrict__ g, const float* __restrict__ beta,
               float* __restrict__ Y)
{
  __shared__ float tile[16][132];            // pitch 132: conflict-free LN scan
  const int ln = threadIdx.x;                // 0..31 (one wave, EXEC all-1s)
  const int lr = ln & 15, lh = ln >> 4;
  const int r0 = blockIdx.x * 16;

  // Preload A fragments: K=128 -> 4 chunks of 32.
  v16h afr[4];
#pragma unroll
  for (int kc = 0; kc < 4; ++kc) {
#pragma unroll
    for (int e = 0; e < 16; ++e) {
      const int k = kc * 32 + 16 * (e >> 3) + 8 * lh + (e & 7);
      afr[kc][e] = (_Float16)X[(r0 + lr) * DD + k];
    }
  }

#pragma unroll
  for (int nc = 0; nc < 8; ++nc) {
    v8f acc = {0.f, 0.f, 0.f, 0.f, 0.f, 0.f, 0.f, 0.f};
#pragma unroll
    for (int kc = 0; kc < 4; ++kc) {
      v16h bfr;
      const int n = nc * 16 + lr;
#pragma unroll
      for (int e = 0; e < 16; ++e) {
        const int k = kc * 32 + 16 * (e >> 3) + 8 * lh + (e & 7);
        bfr[e] = (_Float16)W[n * DD + k];
      }
      acc = wmma16x16x32(afr[kc], bfr, acc);
    }
#pragma unroll
    for (int r = 0; r < 8; ++r)
      tile[r + 8 * lh][nc * 16 + lr] = acc[r];
  }
  __syncthreads();

  if (ln < 16) {
    float mu = 0.f;
    for (int d = 0; d < DD; ++d) mu += tile[ln][d];
    mu *= (1.0f / DD);
    float var = 0.f;
    for (int d = 0; d < DD; ++d) { const float t = tile[ln][d] - mu; var += t * t; }
    var *= (1.0f / DD);
    const float rs = rsqrtf(var + kLnEps);
    for (int d = 0; d < DD; ++d)
      Y[(size_t)(r0 + ln) * DD + d] = (tile[ln][d] - mu) * rs * g[d] + beta[d];
  }
}

// ---------------------------------------------------------------------------
// K2: phases + cos/sin (f16, packed [cos|sin] K=256 rows) + f16 copies of V.
// ---------------------------------------------------------------------------
__global__ __launch_bounds__(256)
void k_phase(const float* __restrict__ qr, const float* __restrict__ qi,
             const float* __restrict__ kr, const float* __restrict__ ki,
             const float* __restrict__ vr, const float* __restrict__ vi,
             float* __restrict__ qp, float* __restrict__ kp,
             _Float16* __restrict__ csq, _Float16* __restrict__ csk,
             _Float16* __restrict__ vrh, _Float16* __restrict__ vih)
{
  const int i = blockIdx.x * blockDim.x + threadIdx.x;
  if (i >= BL * DD) return;
  const int row = i >> 7, col = i & 127;

  const float pq = atan2f(qi[i] + kPhEps, qr[i] + kPhEps);
  const float pk = atan2f(ki[i] + kPhEps, kr[i] + kPhEps);
  qp[i] = pq;
  kp[i] = pk;
  csq[row * 256 + col]       = (_Float16)cosf(pq);
  csq[row * 256 + 128 + col] = (_Float16)sinf(pq);
  csk[row * 256 + col]       = (_Float16)cosf(pk);
  csk[row * 256 + 128 + col] = (_Float16)sinf(pk);
  vrh[i] = (_Float16)vr[i];
  vih[i] = (_Float16)vi[i];
}

// ---------------------------------------------------------------------------
// K3: 16x16 score tile per wave.
//   interference via WMMA (K=256 over [cos|sin]); coherence = exp(-L1) via
//   VALU over LDS phase tiles (pitch 132 -> bank stride 4, conflict-free).
// ---------------------------------------------------------------------------
__global__ __launch_bounds__(32)
void k_scores(const float* __restrict__ qp, const float* __restrict__ kp,
              const _Float16* __restrict__ csq, const _Float16* __restrict__ csk,
              float* __restrict__ S)
{
  __shared__ float qt[16][132];
  __shared__ float kt[16][132];
  const int blk = blockIdx.x;                 // [b(2)][qb(32)][kb(32)]
  const int kb = blk & 31, qb = (blk >> 5) & 31, b = blk >> 10;
  const int q0 = qb * 16, k0 = kb * 16;
  const int ln = threadIdx.x, lr = ln & 15, lh = ln >> 4;

  for (int idx = ln; idx < 16 * DD; idx += 32) {
    const int rr = idx >> 7, dd = idx & 127;
    qt[rr][dd] = qp[(size_t)(b * LL + q0 + rr) * DD + dd];
    kt[rr][dd] = kp[(size_t)(b * LL + k0 + rr) * DD + dd];
  }
  __syncthreads();

  v8f acc = {0.f, 0.f, 0.f, 0.f, 0.f, 0.f, 0.f, 0.f};
#pragma unroll
  for (int kc = 0; kc < 8; ++kc) {            // K = 256
    v16h a, bm;
#pragma unroll
    for (int e = 0; e < 16; ++e) {
      const int kk = kc * 32 + 16 * (e >> 3) + 8 * lh + (e & 7);
      a[e] = csq[(size_t)(b * LL + q0 + lr) * 256 + kk];
    }
#pragma unroll
    for (int e = 0; e < 16; ++e) {
      const int kk = kc * 32 + 16 * (e >> 3) + 8 * lh + (e & 7);
      bm[e] = csk[(size_t)(b * LL + k0 + lr) * 256 + kk];
    }
    acc = wmma16x16x32(a, bm, acc);
  }

  const float scale    = 0.08838834764831845f;  // 128^-0.5
  const float inv_temp = 10.0f;                 // 1/TEMP
#pragma unroll
  for (int r = 0; r < 8; ++r) {
    const int m = r + 8 * lh, n = lr;
    float l1 = 0.f;
    for (int d = 0; d < DD; ++d) l1 += fabsf(qt[m][d] - kt[n][d]);
    const float val = (acc[r] * scale) * __expf(-l1) * inv_temp;
    S[((size_t)(b * LL + q0 + m)) * LL + (k0 + n)] = val;
  }
}

// ---------------------------------------------------------------------------
// K4: row softmax over 512 keys, one wave per row; output f16 attn.
// ---------------------------------------------------------------------------
__global__ __launch_bounds__(32)
void k_softmax(const float* __restrict__ S, _Float16* __restrict__ A)
{
  const int row = blockIdx.x;                 // b*L + q
  const int ln  = threadIdx.x;
  const float* s = S + (size_t)row * LL;

  float v[16];
  float m = -3.4e38f;
#pragma unroll
  for (int j = 0; j < 16; ++j) { v[j] = s[ln + 32 * j]; m = fmaxf(m, v[j]); }
#pragma unroll
  for (int o = 16; o > 0; o >>= 1) m = fmaxf(m, __shfl_xor(m, o, 32));

  float sum = 0.f;
#pragma unroll
  for (int j = 0; j < 16; ++j) { v[j] = __expf(v[j] - m); sum += v[j]; }
#pragma unroll
  for (int o = 16; o > 0; o >>= 1) sum += __shfl_xor(sum, o, 32);

  const float inv = 1.0f / sum;
  _Float16* a = A + (size_t)row * LL;
#pragma unroll
  for (int j = 0; j < 16; ++j) a[ln + 32 * j] = (_Float16)(v[j] * inv);
}

// ---------------------------------------------------------------------------
// K5: out = LayerNorm(attn @ V), 16 q-rows x 128 cols per wave, K=512.
// ---------------------------------------------------------------------------
__global__ __launch_bounds__(32)
void k_av_ln(const _Float16* __restrict__ At, const _Float16* __restrict__ Vh,
             const float* __restrict__ g, const float* __restrict__ beta,
             float* __restrict__ Y)
{
  __shared__ float tile[16][132];
  const int blk = blockIdx.x;                 // [b(2)][qb(32)]
  const int qb = blk & 31, b = blk >> 5;
  const int q0 = qb * 16;
  const int ln = threadIdx.x, lr = ln & 15, lh = ln >> 4;

  v8f acc[8];
#pragma unroll
  for (int nc = 0; nc < 8; ++nc)
    acc[nc] = v8f{0.f, 0.f, 0.f, 0.f, 0.f, 0.f, 0.f, 0.f};

  for (int kc = 0; kc < 16; ++kc) {           // K = 512
    v16h a;
#pragma unroll
    for (int e = 0; e < 16; ++e) {
      const int kk = kc * 32 + 16 * (e >> 3) + 8 * lh + (e & 7);
      a[e] = At[((size_t)(b * LL + q0 + lr)) * LL + kk];
    }
    if (kc + 1 < 16)
      __builtin_prefetch(&At[((size_t)(b * LL + q0 + lr)) * LL + (kc + 1) * 32], 0, 3);
#pragma unroll
    for (int nc = 0; nc < 8; ++nc) {
      v16h bm;
#pragma unroll
      for (int e = 0; e < 16; ++e) {
        const int kk = kc * 32 + 16 * (e >> 3) + 8 * lh + (e & 7);
        bm[e] = Vh[(size_t)(b * LL + kk) * DD + nc * 16 + lr];
      }
      acc[nc] = wmma16x16x32(a, bm, acc[nc]);
    }
  }

#pragma unroll
  for (int nc = 0; nc < 8; ++nc)
#pragma unroll
    for (int r = 0; r < 8; ++r)
      tile[r + 8 * lh][nc * 16 + lr] = acc[nc][r];
  __syncthreads();

  if (ln < 16) {
    float mu = 0.f;
    for (int d = 0; d < DD; ++d) mu += tile[ln][d];
    mu *= (1.0f / DD);
    float var = 0.f;
    for (int d = 0; d < DD; ++d) { const float t = tile[ln][d] - mu; var += t * t; }
    var *= (1.0f / DD);
    const float rs = rsqrtf(var + kLnEps);
    for (int d = 0; d < DD; ++d)
      Y[((size_t)(b * LL + q0 + ln)) * DD + d] = (tile[ln][d] - mu) * rs * g[d] + beta[d];
  }
}

// ---------------------------------------------------------------------------
extern "C" void kernel_launch(void* const* d_in, const int* in_sizes, int n_in,
                              void* d_out, int out_size, void* d_ws, size_t ws_size,
                              hipStream_t stream)
{
  (void)in_sizes; (void)n_in; (void)out_size; (void)ws_size;

  const float* q_real = (const float*)d_in[0];
  const float* q_imag = (const float*)d_in[1];
  const float* k_real = (const float*)d_in[2];
  const float* k_imag = (const float*)d_in[3];
  const float* v_real = (const float*)d_in[4];
  const float* v_imag = (const float*)d_in[5];
  const float* Wq     = (const float*)d_in[6];
  const float* Wk     = (const float*)d_in[7];
  const float* Wv     = (const float*)d_in[8];
  const float* qn_g   = (const float*)d_in[9];
  const float* qn_b   = (const float*)d_in[10];
  const float* kn_g   = (const float*)d_in[11];
  const float* kn_b   = (const float*)d_in[12];
  const float* vn_g   = (const float*)d_in[13];
  const float* vn_b   = (const float*)d_in[14];
  const float* on_g   = (const float*)d_in[15];
  const float* on_b   = (const float*)d_in[16];

  // Workspace carve-out (~8.6 MB, 256B-aligned slices).
  char* ws = (char*)d_ws;
  size_t off = 0;
  auto carve = [&](size_t bytes) -> char* {
    char* p = ws + off;
    off += (bytes + 255) & ~(size_t)255;
    return p;
  };
  float*    qr    = (float*)carve((size_t)BL * DD * 4);
  float*    qi    = (float*)carve((size_t)BL * DD * 4);
  float*    kr    = (float*)carve((size_t)BL * DD * 4);
  float*    ki    = (float*)carve((size_t)BL * DD * 4);
  float*    vr    = (float*)carve((size_t)BL * DD * 4);
  float*    vi    = (float*)carve((size_t)BL * DD * 4);
  float*    qp    = (float*)carve((size_t)BL * DD * 4);
  float*    kp    = (float*)carve((size_t)BL * DD * 4);
  float*    S     = (float*)carve((size_t)BB * LL * LL * 4);
  _Float16* csq   = (_Float16*)carve((size_t)BL * 256 * 2);
  _Float16* csk   = (_Float16*)carve((size_t)BL * 256 * 2);
  _Float16* vrh   = (_Float16*)carve((size_t)BL * DD * 2);
  _Float16* vih   = (_Float16*)carve((size_t)BL * DD * 2);
  _Float16* attnh = (_Float16*)carve((size_t)BB * LL * LL * 2);

  float* out_r = (float*)d_out;
  float* out_i = (float*)d_out + (size_t)BL * DD;

  // 1) Six projection + LayerNorm GEMMs (WMMA f16, f32 accumulate).
  k_proj_ln<<<BL / 16, 32, 0, stream>>>(q_real, Wq, qn_g, qn_b, qr);
  k_proj_ln<<<BL / 16, 32, 0, stream>>>(q_imag, Wq, qn_g, qn_b, qi);
  k_proj_ln<<<BL / 16, 32, 0, stream>>>(k_real, Wk, kn_g, kn_b, kr);
  k_proj_ln<<<BL / 16, 32, 0, stream>>>(k_imag, Wk, kn_g, kn_b, ki);
  k_proj_ln<<<BL / 16, 32, 0, stream>>>(v_real, Wv, vn_g, vn_b, vr);
  k_proj_ln<<<BL / 16, 32, 0, stream>>>(v_imag, Wv, vn_g, vn_b, vi);

  // 2) Phases + cos/sin packing + f16 V copies.
  k_phase<<<(BL * DD + 255) / 256, 256, 0, stream>>>(qr, qi, kr, ki, vr, vi,
                                                     qp, kp, csq, csk, vrh, vih);

  // 3) Scores: WMMA interference (K=256) fused with exp(-L1) coherence.
  k_scores<<<BB * 32 * 32, 32, 0, stream>>>(qp, kp, csq, csk, S);

  // 4) Softmax over keys.
  k_softmax<<<BB * LL, 32, 0, stream>>>(S, attnh);

  // 5) attn @ V + output LayerNorm (WMMA, K=512), real then imag.
  k_av_ln<<<BB * 32, 32, 0, stream>>>(attnh, vrh, on_g, on_b, out_r);
  k_av_ln<<<BB * 32, 32, 0, stream>>>(attnh, vih, on_g, on_b, out_i);
}
